// GatedGCNGraphGymLayer_13005160972635
// MI455X (gfx1250) — compile-verified, matrix operand order checked by reading, not run on hardware
//
#include <hip/hip_runtime.h>

typedef __attribute__((ext_vector_type(16))) __bf16 v16bf;
typedef __attribute__((ext_vector_type(8)))  __bf16 v8bf;
typedef __attribute__((ext_vector_type(4)))  __bf16 v4bf;
typedef __attribute__((ext_vector_type(8)))  float  v8f;

#define N_NODES 40000
#define N_EDGES 640000
#define DD      128
#define BN_EPS  1e-5f
#define AGG_EPS 1e-6f

// ---------------- fragment helpers ----------------

// A-fragment (16x32 bf16) from an LDS row-major bf16 tile: lane L -> row M=L%16,
// K = kb + {0..7,16..23} (lanes 0-15) / kb + {8..15,24..31} (lanes 16-31).
// rowp = &tile[m*ld + kb]; two 16B ds loads.
__device__ __forceinline__ v16bf ldA_lds(const __bf16* rowp, int lane) {
  const int kh = (lane < 16) ? 0 : 8;
  v8bf lo = *(const v8bf*)(rowp + kh);
  v8bf hi = *(const v8bf*)(rowp + kh + 16);
  return __builtin_shufflevector(lo, hi, 0, 1, 2, 3, 4, 5, 6, 7,
                                 8, 9, 10, 11, 12, 13, 14, 15);
}

// B-fragment from pre-swizzled packed bf16 weights: fragment `frag`, lane-contiguous.
__device__ __forceinline__ v16bf ldB_packed(const __bf16* __restrict__ p, int frag, int lane) {
  return *(const v16bf*)(p + ((size_t)frag * 32 + lane) * 16);
}

__device__ __forceinline__ v4bf cvt4(float4 v) {
  v4bf r; r[0] = (__bf16)v.x; r[1] = (__bf16)v.y; r[2] = (__bf16)v.z; r[3] = (__bf16)v.w;
  return r;
}

// ---------------- kernel: repack f32 row-major W[K x N] -> bf16 B-fragment layout ----------------
// One wave per (kt, nt) fragment. Done once; output reused by every GEMM tile.
__global__ void __launch_bounds__(32)
repack_kernel(const float* __restrict__ W, int N, __bf16* __restrict__ out) {
  const int lane = threadIdx.x;
  const int kt = blockIdx.x, nt = blockIdx.y;
  const int Nt = N >> 4;
  const int kh = (lane < 16) ? 0 : 8;
  const int col = nt * 16 + (lane & 15);
  v16bf f;
#pragma unroll
  for (int e = 0; e < 8; ++e) f[e] = (__bf16)W[(size_t)(kt * 32 + kh + e) * N + col];
#pragma unroll
  for (int e = 0; e < 8; ++e) f[8 + e] = (__bf16)W[(size_t)(kt * 32 + kh + 16 + e) * N + col];
  *(v16bf*)(out + ((size_t)(kt * Nt + nt) * 32 + lane) * 16) = f;
}

// ---------------- kernel: zero scratch ----------------
__global__ void __launch_bounds__(256)
zero_kernel(float* __restrict__ p, size_t n) {
  size_t stride = (size_t)gridDim.x * blockDim.x;
  for (size_t i = (size_t)blockIdx.x * blockDim.x + threadIdx.x; i < n; i += stride)
    p[i] = 0.f;
}

// ---------------- kernel: node GEMMs Ax/Bx/Dx/Ex ----------------
// Block = 256 threads (8 waves) = one 16-row tile x all 8 N-tiles; blockIdx.y picks weight.
__global__ void __launch_bounds__(256)
node_gemm_kernel(const float* __restrict__ x,
                 const __bf16* __restrict__ pAw, const float* __restrict__ Ab,
                 const __bf16* __restrict__ pBw, const float* __restrict__ Bb,
                 const __bf16* __restrict__ pDw, const float* __restrict__ Db,
                 const __bf16* __restrict__ pEw, const float* __restrict__ Eb,
                 float* __restrict__ Ax, float* __restrict__ Bx,
                 float* __restrict__ Dx, float* __restrict__ Ex) {
  __shared__ __align__(16) __bf16 tile[16 * DD];
  const int tid = threadIdx.x;
  const int m0 = blockIdx.x * 16;
  const __bf16* pW; const float* bias; float* out;
  switch (blockIdx.y) {
    case 0:  pW = pAw; bias = Ab; out = Ax; break;
    case 1:  pW = pBw; bias = Bb; out = Bx; break;
    case 2:  pW = pDw; bias = Db; out = Dx; break;
    default: pW = pEw; bias = Eb; out = Ex; break;
  }
  // stage x tile (16x128 f32 -> bf16): 512 float4 items, 2 per thread
#pragma unroll
  for (int j = 0; j < 2; ++j) {
    const int item = tid + j * 256;
    const int row = item >> 5;
    const int col = (item & 31) * 4;
    *(v4bf*)(&tile[row * DD + col]) = cvt4(*(const float4*)(x + (size_t)(m0 + row) * DD + col));
  }
  __syncthreads();

  const int lane = tid & 31;
  const int wv = tid >> 5;                  // N-tile 0..7
  const int n = wv * 16 + (lane & 15);
  v8f c = {};
#pragma unroll
  for (int ki = 0; ki < 4; ++ki) {
    v16bf a = ldA_lds(&tile[(lane & 15) * DD + ki * 32], lane);
    v16bf b = ldB_packed(pW, ki * 8 + wv, lane);
    c = __builtin_amdgcn_wmma_f32_16x16x32_bf16(false, a, false, b, (short)0, c, false, false);
  }
  const float bv = bias[n];
  const int rbase = (lane < 16) ? 0 : 8;
#pragma unroll
  for (int r = 0; r < 8; ++r)
    out[(size_t)(m0 + rbase + r) * DD + n] = c[r] + bv;
}

// ---------------- kernel: Ce GEMM + message + sigmoid + atomic scatter ----------------
__global__ void __launch_bounds__(256)
edge_msg_kernel(const float* __restrict__ e, const __bf16* __restrict__ pCw,
                const float* __restrict__ Cb,
                const int* __restrict__ srcI, const int* __restrict__ dstI,
                const float* __restrict__ Bx, const float* __restrict__ Dx,
                const float* __restrict__ Ex,
                float* __restrict__ num, float* __restrict__ den,
                float* __restrict__ e_pre) {
  __shared__ __align__(16) __bf16 tile[16 * DD];
  const int tid = threadIdx.x;
  const int m0 = blockIdx.x * 16;
#pragma unroll
  for (int j = 0; j < 2; ++j) {
    const int item = tid + j * 256;
    const int row = item >> 5;
    const int col = (item & 31) * 4;
    *(v4bf*)(&tile[row * DD + col]) = cvt4(*(const float4*)(e + (size_t)(m0 + row) * DD + col));
  }
  __syncthreads();

  const int lane = tid & 31;
  const int wv = tid >> 5;                  // N-tile 0..7
  const int n = wv * 16 + (lane & 15);
  v8f c = {};
#pragma unroll
  for (int ki = 0; ki < 4; ++ki) {
    v16bf a = ldA_lds(&tile[(lane & 15) * DD + ki * 32], lane);
    v16bf b = ldB_packed(pCw, ki * 8 + wv, lane);
    c = __builtin_amdgcn_wmma_f32_16x16x32_bf16(false, a, false, b, (short)0, c, false, false);
  }
  const float bv = Cb[n];
  const int rbase = (lane < 16) ? 0 : 8;
#pragma unroll
  for (int r = 0; r < 8; ++r) {
    const int edge = m0 + rbase + r;
    const int s = srcI[edge], d = dstI[edge];
    const float v = c[r] + bv + Dx[(size_t)d * DD + n] + Ex[(size_t)s * DD + n];
    e_pre[(size_t)edge * DD + n] = v;       // pre-BN e_out straight into d_out
    const float sg = 1.f / (1.f + __expf(-v));
    atomicAdd(&num[(size_t)d * DD + n], sg * Bx[(size_t)s * DD + n]);
    atomicAdd(&den[(size_t)d * DD + n], sg);
  }
}

// ---------------- kernel: x_pre = Ax + num/(den+eps) + BN partials ----------------
__global__ void __launch_bounds__(128)
x_finalize_kernel(const float* __restrict__ Ax, const float* __restrict__ num,
                  const float* __restrict__ den, float* __restrict__ x_pre,
                  float* __restrict__ xsum, float* __restrict__ xsq) {
  const int c = threadIdx.x;
  const int r0 = blockIdx.x * 64;
  float s = 0.f, q = 0.f;
  for (int r = r0; r < r0 + 64; ++r) {
    const size_t i = (size_t)r * DD + c;
    const float v = Ax[i] + num[i] / (den[i] + AGG_EPS);
    x_pre[i] = v;
    s += v; q += v * v;
  }
  atomicAdd(&xsum[c], s);
  atomicAdd(&xsq[c], q);
}

// ---------------- kernel: column sum/sumsq ----------------
__global__ void __launch_bounds__(128)
col_stats_kernel(const float* __restrict__ m, int rows_per_block,
                 float* __restrict__ sum, float* __restrict__ sq) {
  const int c = threadIdx.x;
  const int r0 = blockIdx.x * rows_per_block;
  float s = 0.f, q = 0.f;
  for (int r = r0; r < r0 + rows_per_block; ++r) {
    const float v = m[(size_t)r * DD + c];
    s += v; q += v * v;
  }
  atomicAdd(&sum[c], s);
  atomicAdd(&sq[c], q);
}

// ---------------- kernel: in-place BatchNorm (batch stats) + ReLU ----------------
__global__ void __launch_bounds__(256)
bn_apply_kernel(float* __restrict__ h, size_t nelem, float invN,
                const float* __restrict__ sum, const float* __restrict__ sq,
                const float* __restrict__ g, const float* __restrict__ b) {
  size_t stride = (size_t)gridDim.x * blockDim.x;
  for (size_t i = (size_t)blockIdx.x * blockDim.x + threadIdx.x; i < nelem; i += stride) {
    const int c = (int)(i & (DD - 1));
    const float mu = sum[c] * invN;
    const float var = sq[c] * invN - mu * mu;
    const float v = (h[i] - mu) * rsqrtf(var + BN_EPS) * g[c] + b[c];
    h[i] = fmaxf(v, 0.f);
  }
}

// ---------------- kernel: edge MLP ----------------
// Block = 512 threads (16 waves) = 16 edges x 256 hidden; feat tile staged once in LDS.
__global__ void __launch_bounds__(512)
edge_mlp_kernel(const float* __restrict__ xo, const float* __restrict__ eo,
                const int* __restrict__ srcI, const int* __restrict__ dstI,
                const __bf16* __restrict__ pW1, const float* __restrict__ b1,
                const float* __restrict__ w2, const float* __restrict__ b2,
                float* __restrict__ scores) {
  __shared__ __align__(16) __bf16 tile[16 * 384];
  __shared__ float sc[16];
  const int tid = threadIdx.x;
  const int et = blockIdx.x;
  if (tid < 16) sc[tid] = 0.f;

  // stage feat tile [16 x 384] = cat(xo[src], xo[dst], eo): 1536 float4 items, 3/thread
#pragma unroll
  for (int j = 0; j < 3; ++j) {
    const int item = tid + j * 512;
    const int row = item / 96;
    const int col = (item % 96) * 4;
    const int edge = et * 16 + row;
    const float* p;
    if (col < 128)      p = xo + (size_t)srcI[edge] * DD + col;
    else if (col < 256) p = xo + (size_t)dstI[edge] * DD + (col - 128);
    else                p = eo + (size_t)edge * DD + (col - 256);
    *(v4bf*)(&tile[row * 384 + col]) = cvt4(*(const float4*)p);
  }
  __syncthreads();

  const int lane = tid & 31;
  const int wv = tid >> 5;                  // N-tile 0..15
  const int n = wv * 16 + (lane & 15);
  v8f c = {};
#pragma unroll
  for (int ki = 0; ki < 12; ++ki) {
    v16bf a = ldA_lds(&tile[(lane & 15) * 384 + ki * 32], lane);
    v16bf b = ldB_packed(pW1, ki * 16 + wv, lane);
    c = __builtin_amdgcn_wmma_f32_16x16x32_bf16(false, a, false, b, (short)0, c, false, false);
  }

  const float bias = b1[n];
  const float w2c = w2[n];
  const int rbase = (lane < 16) ? 0 : 8;
#pragma unroll
  for (int r = 0; r < 8; ++r) {
    float h = fmaxf(c[r] + bias, 0.f);
    float v = h * w2c;
#pragma unroll
    for (int off = 1; off < 16; off <<= 1) v += __shfl_xor(v, off, 32);
    if ((lane & 15) == 0) atomicAdd(&sc[rbase + r], v);
  }
  __syncthreads();
  if (tid < 16) {
    const float v = sc[tid] + b2[0];
    scores[(size_t)et * 16 + tid] = 1.f / (1.f + __expf(-v));
  }
}

extern "C" void kernel_launch(void* const* d_in, const int* in_sizes, int n_in,
                              void* d_out, int out_size, void* d_ws, size_t ws_size,
                              hipStream_t stream) {
  (void)in_sizes; (void)n_in; (void)out_size; (void)ws_size;
  const float* x  = (const float*)d_in[0];
  const float* e  = (const float*)d_in[1];
  const int*   ei = (const int*)d_in[2];
  const int* srcI = ei;            // edge_index[0] = source j
  const int* dstI = ei + N_EDGES;  // edge_index[1] = target i
  const float *Aw = (const float*)d_in[3],  *Ab = (const float*)d_in[4];
  const float *Bw = (const float*)d_in[5],  *Bb = (const float*)d_in[6];
  const float *Cw = (const float*)d_in[7],  *Cb = (const float*)d_in[8];
  const float *Dw = (const float*)d_in[9],  *Db = (const float*)d_in[10];
  const float *Ew = (const float*)d_in[11], *Eb = (const float*)d_in[12];
  const float *bnxg = (const float*)d_in[13], *bnxb = (const float*)d_in[14];
  const float *bneg = (const float*)d_in[15], *bneb = (const float*)d_in[16];
  const float *W1 = (const float*)d_in[17], *b1 = (const float*)d_in[18];
  const float *w2 = (const float*)d_in[19], *b2 = (const float*)d_in[20];

  // d_out layout: x_out [40000*128] | e_out [640000*128] | scores [640000]
  float* out    = (float*)d_out;
  float* x_out  = out;
  float* e_out  = out + (size_t)N_NODES * DD;
  float* scores = e_out + (size_t)N_EDGES * DD;

  // ws layout (floats): Ax|Bx|Dx|Ex|num|den|stats(512)| packed bf16 weights (~352 KB)
  const size_t nN = (size_t)N_NODES * DD;
  float* ws   = (float*)d_ws;
  float* Ax   = ws;
  float* Bx   = Ax + nN;
  float* Dx   = Bx + nN;
  float* Ex   = Dx + nN;
  float* num  = Ex + nN;
  float* den  = num + nN;
  float* xsum = den + nN;
  float* xsq  = xsum + DD;
  float* esum = xsq + DD;
  float* esq  = esum + DD;
  __bf16* pAw = (__bf16*)(esq + DD);       // 32B-aligned (offset is a multiple of 32B)
  __bf16* pBw = pAw + 128 * 128;
  __bf16* pCw = pBw + 128 * 128;
  __bf16* pDw = pCw + 128 * 128;
  __bf16* pEw = pDw + 128 * 128;
  __bf16* pW1 = pEw + 128 * 128;           // 384*256

  // 0) one-time weight repack to bf16 WMMA B-fragment layout
  dim3 g128(4, 8);                          // K/32 x N/16 for 128x128
  repack_kernel<<<g128, 32, 0, stream>>>(Aw, 128, pAw);
  repack_kernel<<<g128, 32, 0, stream>>>(Bw, 128, pBw);
  repack_kernel<<<g128, 32, 0, stream>>>(Cw, 128, pCw);
  repack_kernel<<<g128, 32, 0, stream>>>(Dw, 128, pDw);
  repack_kernel<<<g128, 32, 0, stream>>>(Ew, 128, pEw);
  dim3 gW1(12, 16);                         // 384/32 x 256/16
  repack_kernel<<<gW1, 32, 0, stream>>>(W1, 256, pW1);

  // 1) zero num|den|stats (contiguous)
  zero_kernel<<<512, 256, 0, stream>>>(num, 2 * nN + 4 * DD);

  // 2) node GEMMs (bf16 WMMA, LDS-staged A, packed B)
  dim3 gN(N_NODES / 16, 4);
  node_gemm_kernel<<<gN, 256, 0, stream>>>(x, pAw, Ab, pBw, Bb, pDw, Db, pEw, Eb,
                                           Ax, Bx, Dx, Ex);

  // 3) Ce GEMM + message + sigmoid + scatter atomics; pre-BN e_out into d_out
  edge_msg_kernel<<<N_EDGES / 16, 256, 0, stream>>>(e, pCw, Cb, srcI, dstI,
                                                    Bx, Dx, Ex, num, den, e_out);

  // 4) x_out pre-BN + column partials
  x_finalize_kernel<<<N_NODES / 64, 128, 0, stream>>>(Ax, num, den, x_out, xsum, xsq);

  // 5) e_out column partials
  col_stats_kernel<<<N_EDGES / 256, 128, 0, stream>>>(e_out, 256, esum, esq);

  // 6/7) in-place BN + ReLU
  bn_apply_kernel<<<2048, 256, 0, stream>>>(x_out, nN, 1.f / N_NODES, xsum, xsq, bnxg, bnxb);
  bn_apply_kernel<<<4096, 256, 0, stream>>>(e_out, (size_t)N_EDGES * DD, 1.f / N_EDGES,
                                            esum, esq, bneg, bneb);

  // 8) edge scoring MLP (bf16 WMMA over K=384, LDS-staged feat tile)
  edge_mlp_kernel<<<N_EDGES / 16, 512, 0, stream>>>(x_out, e_out, srcI, dstI,
                                                    pW1, b1, w2, b2, scores);
}
